// MDRMWithCPRecon_14688788152420
// MI455X (gfx1250) — matrix operand order repaired and checked by reading, not compiled, not run on hardware
//
#include <hip/hip_runtime.h>

#define BB 8
#define CC 256
#define HH 64
#define WWD 64
#define KR 4
#define HWP (HH*WWD)          // 4096
#define C2 (2*CC)             // 512
#define KDIM (C2*9)           // 4608
#define TOTE (BB*CC*HWP)      // 8388608
#define PW 66
#define PPLANE (PW*PW)        // 4356 (padded 66x66 plane)

typedef __attribute__((ext_vector_type(16))) __bf16 v16bf;
typedef __attribute__((ext_vector_type(8)))  float  v8f;
typedef __attribute__((ext_vector_type(8)))  unsigned short v8us;

__device__ __forceinline__ unsigned short f32_to_bf16(float f) {
  unsigned int u = __float_as_uint(f);
  unsigned int r = 0x7FFFu + ((u >> 16) & 1u);
  return (unsigned short)((u + r) >> 16);
}
__device__ __forceinline__ float sigmoidf_(float x) { return 1.0f / (1.0f + __expf(-x)); }

// ---------------------------------------------------------------------------
// Kernel 0a: zero-padded bf16 input: xp[b][c(512)][66][66], border = 0
// ---------------------------------------------------------------------------
__global__ __launch_bounds__(256)
void pad_kernel(const float* __restrict__ frm, const float* __restrict__ oth,
                unsigned short* __restrict__ xp) {
  int idx = blockIdx.x * 256 + threadIdx.x;
  if (idx >= BB * C2 * PPLANE) return;
  int p  = idx % PPLANE;
  int bc = idx / PPLANE;
  int y = p / PW, x = p - y * PW;
  unsigned short v = 0;
  if (y >= 1 && y <= 64 && x >= 1 && x <= 64) {
    int c = bc & (C2 - 1);
    int b = bc >> 9;
    const float* src = (c < CC) ? frm : oth;
    v = f32_to_bf16(src[((size_t)(b * CC + (c & (CC - 1))) << 12) + ((y - 1) << 6) + (x - 1)]);
  }
  xp[idx] = v;
}

// ---------------------------------------------------------------------------
// Kernel 0b: transpose + convert weights: W3[o][c][tap] f32 -> Wtr[o][tap][c] bf16
// ---------------------------------------------------------------------------
__global__ __launch_bounds__(256)
void wtrans_kernel(const float* __restrict__ W3, unsigned short* __restrict__ Wtr) {
  int idx = blockIdx.x * 256 + threadIdx.x;
  if (idx >= CC * 9 * C2) return;
  int o = idx / (9 * C2);
  int rem = idx - o * 9 * C2;
  int tap = rem >> 9;
  int c   = rem & (C2 - 1);
  Wtr[idx] = f32_to_bf16(W3[(size_t)o * KDIM + c * 9 + tap]);
}

// ---------------------------------------------------------------------------
// Kernel 1: tap-decomposed implicit-GEMM 3x3 conv + bias + leaky_relu.
// Block = 256 thr (8 waves), tile = 64 pixels (one image row) x 128 out-chan.
// Waves 4x2; each wave: 1 A-fragment, 4 accumulators -> 4 WMMAs per K-chunk.
// A: 8 unconditional u16 loads from padded bf16 planes -> 1 ds_store_b128.
// B: 2 async global->LDS b128 copies per thread (ASYNCcnt path).
// ---------------------------------------------------------------------------
__global__ __launch_bounds__(256)
void conv3x3_wmma(const unsigned short* __restrict__ xp,
                  const unsigned short* __restrict__ Wtr,
                  const float* __restrict__ b3, float* __restrict__ Fm) {
  __shared__ unsigned short Asub[64][32];    // [w-pixel][c-chunk]  4 KB
  __shared__ unsigned short Bsub[128][32];   // [out-chan][c-chunk] 8 KB

  const int tid  = threadIdx.x;
  const int lane = tid & 31;
  const int wave = tid >> 5;
  const int wm   = wave & 3;        // M subtile (16 pixels)
  const int wn   = wave >> 2;       // N half (64 channels -> 4 accs)
  const int hrow = blockIdx.x;      // image row
  const int n0g  = blockIdx.y * 128;
  const int b    = blockIdx.z;

  // A loader: pixel column am, 8 channels starting at ak0
  const int am  = tid & 63;
  const int ak0 = (tid >> 6) * 8;
  // B loader: two async b128 per thread
  const int bn0 = tid >> 2;          // 0..63
  const int bn1 = bn0 + 64;          // 64..127
  const int bk0 = (tid & 3) * 8;
  const unsigned ldsB0 = (unsigned)(unsigned long long)(const void*)&Bsub[bn0][bk0];
  const unsigned ldsB1 = (unsigned)(unsigned long long)(const void*)&Bsub[bn1][bk0];

  v8f acc0 = {}, acc1 = {}, acc2 = {}, acc3 = {};
  const int half = lane >> 4;
  const int l15  = lane & 15;
  const v8us* arow  = (const v8us*)(&Asub[wm * 16 + l15][0]);
  const v8us* brow0 = (const v8us*)(&Bsub[wn * 64 +  0 + l15][0]);
  const v8us* brow1 = (const v8us*)(&Bsub[wn * 64 + 16 + l15][0]);
  const v8us* brow2 = (const v8us*)(&Bsub[wn * 64 + 32 + l15][0]);
  const v8us* brow3 = (const v8us*)(&Bsub[wn * 64 + 48 + l15][0]);

  // base of this thread's 8 padded input planes (channel = ak0 + j + c0)
  const unsigned short* xbase = xp + (size_t)(b * C2 + ak0) * PPLANE;

  for (int tap = 0; tap < 9; ++tap) {
    const int r = tap / 3;
    const int s = tap - r * 3;
    const int tapoff = (hrow + r) * PW + (am + s);   // padded-plane offset, never OOB
    const unsigned short* w0 = Wtr + ((size_t)(n0g + bn0) * 9 + tap) * (size_t)C2 + bk0;
    const unsigned short* w1 = Wtr + ((size_t)(n0g + bn1) * 9 + tap) * (size_t)C2 + bk0;

    for (int ct = 0; ct < C2 / 32; ++ct) {
      const int c0 = ct * 32;
      __syncthreads();
      // ---- B tile: two async bf16 weight copies into LDS ----
      asm volatile("global_load_async_to_lds_b128 %0, %1, off"
                   :: "v"(ldsB0), "v"((unsigned long long)(w0 + c0)) : "memory");
      asm volatile("global_load_async_to_lds_b128 %0, %1, off"
                   :: "v"(ldsB1), "v"((unsigned long long)(w1 + c0)) : "memory");
      // ---- A tile: 8 unconditional u16 loads (plane-strided), pack, 16B store ----
      const unsigned short* ap = xbase + (size_t)c0 * PPLANE + tapoff;
      v8us pk;
      #pragma unroll
      for (int j = 0; j < 8; ++j) pk[j] = ap[j * PPLANE];
      *(v8us*)(&Asub[am][ak0]) = pk;
      asm volatile("s_wait_asynccnt 0" ::: "memory");
      __syncthreads();

      // ---- fragments: A(16x32)/B(32x16) in contiguous k order, k = 16*half + s ----
      union { v8us h[2]; v16bf v; } ua, ub0, ub1, ub2, ub3;
      ua.h[0]  = arow[half * 2];
      ua.h[1]  = arow[half * 2 + 1];
      ub0.h[0] = brow0[half * 2]; ub0.h[1] = brow0[half * 2 + 1];
      ub1.h[0] = brow1[half * 2]; ub1.h[1] = brow1[half * 2 + 1];
      ub2.h[0] = brow2[half * 2]; ub2.h[1] = brow2[half * 2 + 1];
      ub3.h[0] = brow3[half * 2]; ub3.h[1] = brow3[half * 2 + 1];
      acc0 = __builtin_amdgcn_wmma_f32_16x16x32_bf16(false, ua.v, false, ub0.v, (short)0, acc0, false, false);
      acc1 = __builtin_amdgcn_wmma_f32_16x16x32_bf16(false, ua.v, false, ub1.v, (short)0, acc1, false, false);
      acc2 = __builtin_amdgcn_wmma_f32_16x16x32_bf16(false, ua.v, false, ub2.v, (short)0, acc2, false, false);
      acc3 = __builtin_amdgcn_wmma_f32_16x16x32_bf16(false, ua.v, false, ub3.v, (short)0, acc3, false, false);
    }
  }

  // ---- epilogue: C layout VGPR rr -> M = half*8 + rr, N = l15 ----
  const int pb  = hrow * 64 + wm * 16 + half * 8;
  const int oc  = n0g + wn * 64 + l15;
  v8f accs[4] = {acc0, acc1, acc2, acc3};
  #pragma unroll
  for (int q = 0; q < 4; ++q) {
    const int oq = oc + q * 16;
    const float bias = b3[oq];
    const size_t ob = ((size_t)(b * CC + oq)) << 12;
    #pragma unroll
    for (int rr = 0; rr < 8; ++rr) {
      float v = accs[q][rr] + bias;
      v = (v > 0.f) ? v : 0.01f * v;
      Fm[ob + pb + rr] = v;
    }
  }
}

// ---------------------------------------------------------------------------
// Kernel 2: collapse adapter+Wu to effective 4x2 matrices per mode
// ---------------------------------------------------------------------------
__global__ void combos_kernel(const float* __restrict__ Wu, const float* __restrict__ bu,
                              const float* __restrict__ Wa1, const float* __restrict__ ba1,
                              const float* __restrict__ Wa2, const float* __restrict__ ba2,
                              const float* __restrict__ Wa3, const float* __restrict__ ba3,
                              float* __restrict__ cmb, float* __restrict__ cmbb) {
  int t = threadIdx.x;
  const float* Was[3] = {Wa1, Wa2, Wa3};
  const float* bas[3] = {ba1, ba2, ba3};
  if (t < 24) {
    int m = t / 8, k = (t / 2) & 3, i = t & 1;
    float s = 0.f;
    for (int c = 0; c < CC; ++c) s += Wu[k * CC + c] * Was[m][c * 2 + i];
    cmb[(m * 4 + k) * 2 + i] = s;
  } else if (t < 36) {
    int q = t - 24, m = q / 4, k = q & 3;
    float s = bu[k];
    for (int c = 0; c < CC; ++c) s += Wu[k * CC + c] * bas[m][c];
    cmbb[m * 4 + k] = s;
  }
}

// ---------------------------------------------------------------------------
// Kernel 3: pooled avg/max per mode row + tiny 2->4 linear + softmax
// ---------------------------------------------------------------------------
__global__ __launch_bounds__(256)
void pool_make_u(const float* __restrict__ Fm, const float* __restrict__ cmb,
                 const float* __restrict__ cmbb,
                 float* __restrict__ U1, float* __restrict__ U2, float* __restrict__ U3) {
  __shared__ float ssum[256];
  __shared__ float smax[256];
  const int row = blockIdx.x;
  const int tid = threadIdx.x;

  int mode, b, n, M;
  size_t base;
  if (row < BB * CC)                 { mode = 0; b = row >> 8; n = row & 255; M = HWP;       base = (size_t)row << 12; }
  else if (row < BB * CC + BB * WWD) { int r2 = row - BB * CC; mode = 1; b = r2 >> 6; n = r2 & 63; M = CC * HH;  base = ((size_t)b << 20) + n; }
  else                               { int r3 = row - BB * CC - BB * WWD; mode = 2; b = r3 >> 6; n = r3 & 63; M = CC * WWD; base = ((size_t)b << 20) + ((size_t)n << 6); }

  float s = 0.f, mx = -3.4e38f;
  for (int i = tid; i < M; i += 256) {
    size_t off;
    if (mode == 0)      off = (size_t)i;
    else if (mode == 1) off = (size_t)i << 6;
    else                off = ((size_t)(i >> 6) << 12) + (size_t)(i & 63);
    float v = Fm[base + off];
    s += v; mx = fmaxf(mx, v);
  }
  ssum[tid] = s; smax[tid] = mx;
  __syncthreads();
  for (int st = 128; st > 0; st >>= 1) {
    if (tid < st) { ssum[tid] += ssum[tid + st]; smax[tid] = fmaxf(smax[tid], smax[tid + st]); }
    __syncthreads();
  }
  if (tid == 0) {
    float avg = ssum[0] / (float)M;
    float mv  = smax[0];
    float u[KR], um = -3.4e38f;
    for (int k = 0; k < KR; ++k) {
      u[k] = cmb[(mode * 4 + k) * 2 + 0] * avg + cmb[(mode * 4 + k) * 2 + 1] * mv + cmbb[mode * 4 + k];
      um = fmaxf(um, u[k]);
    }
    float es = 0.f;
    for (int k = 0; k < KR; ++k) { u[k] = __expf(u[k] - um); es += u[k]; }
    float inv = 1.f / es;
    float* U = (mode == 0) ? U1 : ((mode == 1) ? U2 : U3);
    int N = (mode == 0) ? CC : 64;
    for (int k = 0; k < KR; ++k) U[((size_t)b * N + n) * KR + k] = u[k] * inv;
  }
}

// ---------------------------------------------------------------------------
// Kernel 4: spatial[b,h,w] = sigmoid(ws * <U3[b,h], U2[b,w]> + bs)
// ---------------------------------------------------------------------------
__global__ void spatial_kernel(const float* __restrict__ U2, const float* __restrict__ U3,
                               const float* __restrict__ wsp, const float* __restrict__ bsp,
                               float* __restrict__ spatial) {
  int idx = blockIdx.x * blockDim.x + threadIdx.x;
  if (idx >= BB * HWP) return;
  int b = idx >> 12, h = (idx >> 6) & 63, w = idx & 63;
  float d = 0.f;
  #pragma unroll
  for (int r = 0; r < KR; ++r)
    d += U3[((b << 6) + h) * KR + r] * U2[((b << 6) + w) * KR + r];
  spatial[idx] = sigmoidf_(wsp[0] * d + bsp[0]);
}

// ---------------------------------------------------------------------------
// Kernel 5: F_spe pooling -> ga (mean over 128), gm (max over 128)
// ---------------------------------------------------------------------------
__global__ void specpool_kernel(const float* __restrict__ U1, const float* __restrict__ U2,
                                const float* __restrict__ U3,
                                float* __restrict__ ga, float* __restrict__ gm) {
  int idx = blockIdx.x * blockDim.x + threadIdx.x;
  if (idx >= BB * CC) return;
  int b = idx >> 8;
  const float* u1 = U1 + (size_t)idx * KR;
  float s = 0.f, mx = -3.4e38f;
  for (int n = 0; n < 128; ++n) {
    const float* uc = (n < 64) ? (U2 + (size_t)(b * 64 + n) * KR)
                               : (U3 + (size_t)(b * 64 + (n - 64)) * KR);
    float d = 0.f;
    #pragma unroll
    for (int r = 0; r < KR; ++r) d += u1[r] * uc[r];
    s += d; mx = fmaxf(mx, d);
  }
  ga[idx] = s / 128.f;
  gm[idx] = mx;
}

// ---------------------------------------------------------------------------
// Kernel 6: spectral[b,o] = sigmoid(sigmoid(Wsa@ga + bsa + Wsm@gm + bsm))
// ---------------------------------------------------------------------------
__global__ void spectral_kernel(const float* __restrict__ Wsa, const float* __restrict__ bsa,
                                const float* __restrict__ Wsm, const float* __restrict__ bsm,
                                const float* __restrict__ ga, const float* __restrict__ gm,
                                float* __restrict__ spec) {
  int idx = blockIdx.x * blockDim.x + threadIdx.x;
  if (idx >= BB * CC) return;
  int b = idx >> 8, o = idx & 255;
  float sa = bsa[o], sm = bsm[o];
  const float* g1 = ga + (b << 8);
  const float* g2 = gm + (b << 8);
  for (int i = 0; i < CC; ++i) { sa += Wsa[o * CC + i] * g1[i]; sm += Wsm[o * CC + i] * g2[i]; }
  spec[idx] = sigmoidf_(sigmoidf_(sa + sm));
}

// ---------------------------------------------------------------------------
// Kernel 7: Vl[b,o,r] = lam[r] * sum_i Wr[o,i]*U1[b,i,r]   (CP factorization)
// ---------------------------------------------------------------------------
__global__ void vlam_kernel(const float* __restrict__ Wr, const float* __restrict__ U1,
                            const float* __restrict__ lam, float* __restrict__ Vl) {
  int idx = blockIdx.x * blockDim.x + threadIdx.x;
  if (idx >= BB * CC * KR) return;
  int b = idx >> 10, o = (idx >> 2) & 255, r = idx & 3;
  float s = 0.f;
  const float* u1 = U1 + (size_t)(b * CC) * KR + r;
  for (int i = 0; i < CC; ++i) s += Wr[o * CC + i] * u1[i * KR];
  Vl[idx] = s * lam[r];
}

// ---------------------------------------------------------------------------
// Kernel 8: fused output + CP recon epilogue (bandwidth-bound)
// ---------------------------------------------------------------------------
__global__ __launch_bounds__(256)
void final_kernel(const float* __restrict__ frm, const float* __restrict__ oth,
                  const float* __restrict__ Fm, const float* __restrict__ spatial,
                  const float* __restrict__ spec, const float* __restrict__ Vl,
                  const float* __restrict__ U2, const float* __restrict__ U3,
                  const float* __restrict__ br, const float* __restrict__ alpha_p,
                  float* __restrict__ out) {
  size_t idx = (size_t)blockIdx.x * 256 + threadIdx.x;
  if (idx >= (size_t)TOTE) return;
  int b = (int)(idx >> 20);
  int c = (int)(idx >> 12) & 255;
  int hw = (int)idx & 4095;
  int h = hw >> 6, w = hw & 63;
  float alpha = alpha_p[0];
  float wt = spec[(b << 8) + c] * spatial[((size_t)b << 12) + hw];
  float f = frm[idx], o = oth[idx];
  float fused = alpha * wt * f + (1.f - alpha) * (1.f - wt) * o;
  float rec = br[c];
  const float* vl = Vl + ((size_t)(b * CC + c) << 2);
  const float* u2 = U2 + ((size_t)(b * 64 + w) << 2);
  const float* u3 = U3 + ((size_t)(b * 64 + h) << 2);
  #pragma unroll
  for (int r = 0; r < 4; ++r) rec += vl[r] * u2[r] * u3[r];
  out[idx] = fused;
  out[(size_t)TOTE + idx] = rec * wt + Fm[idx];
}

// ---------------------------------------------------------------------------
extern "C" void kernel_launch(void* const* d_in, const int* in_sizes, int n_in,
                              void* d_out, int out_size, void* d_ws, size_t ws_size,
                              hipStream_t stream) {
  const float* frm = (const float*)d_in[0];
  const float* oth = (const float*)d_in[1];
  const float* W3  = (const float*)d_in[2];
  const float* b3  = (const float*)d_in[3];
  const float* Wa1 = (const float*)d_in[4];
  const float* ba1 = (const float*)d_in[5];
  const float* Wa2 = (const float*)d_in[6];
  const float* ba2 = (const float*)d_in[7];
  const float* Wa3 = (const float*)d_in[8];
  const float* ba3 = (const float*)d_in[9];
  const float* Wu  = (const float*)d_in[10];
  const float* bu  = (const float*)d_in[11];
  const float* Wr  = (const float*)d_in[12];
  const float* br  = (const float*)d_in[13];
  const float* wsp = (const float*)d_in[14];
  const float* bsp = (const float*)d_in[15];
  const float* Wsa = (const float*)d_in[16];
  const float* bsa = (const float*)d_in[17];
  const float* Wsm = (const float*)d_in[18];
  const float* bsm = (const float*)d_in[19];
  const float* alp = (const float*)d_in[20];
  const float* lam = (const float*)d_in[21];
  float* out = (float*)d_out;

  float* Fm   = (float*)d_ws;                 // 8.39M floats
  float* sp   = Fm + (size_t)TOTE;            // spatial: 32768
  float* U1   = sp + BB * HWP;                // 8192
  float* U2   = U1 + BB * CC * KR;            // 2048
  float* U3   = U2 + BB * WWD * KR;           // 2048
  float* ga   = U3 + BB * HH * KR;            // 2048
  float* gm   = ga + BB * CC;                 // 2048
  float* spec = gm + BB * CC;                 // 2048
  float* Vl   = spec + BB * CC;               // 8192
  float* cmb  = Vl + BB * CC * KR;            // 24
  float* cmbb = cmb + 24;                     // 12
  unsigned short* Wtr = (unsigned short*)(cmbb + 12);       // 256*9*512 bf16
  unsigned short* xp  = Wtr + (size_t)CC * 9 * C2;          // 8*512*66*66 bf16

  pad_kernel<<<(BB * C2 * PPLANE + 255) / 256, 256, 0, stream>>>(frm, oth, xp);
  wtrans_kernel<<<(CC * 9 * C2 + 255) / 256, 256, 0, stream>>>(W3, Wtr);
  conv3x3_wmma<<<dim3(HH, CC / 128, BB), 256, 0, stream>>>(xp, Wtr, b3, Fm);
  combos_kernel<<<1, 64, 0, stream>>>(Wu, bu, Wa1, ba1, Wa2, ba2, Wa3, ba3, cmb, cmbb);
  pool_make_u<<<BB * CC + BB * WWD + BB * HH, 256, 0, stream>>>(Fm, cmb, cmbb, U1, U2, U3);
  spatial_kernel<<<(BB * HWP + 255) / 256, 256, 0, stream>>>(U2, U3, wsp, bsp, sp);
  specpool_kernel<<<(BB * CC + 255) / 256, 256, 0, stream>>>(U1, U2, U3, ga, gm);
  spectral_kernel<<<(BB * CC + 255) / 256, 256, 0, stream>>>(Wsa, bsa, Wsm, bsm, ga, gm, spec);
  vlam_kernel<<<(BB * CC * KR + 255) / 256, 256, 0, stream>>>(Wr, U1, lam, Vl);
  final_kernel<<<(TOTE + 255) / 256, 256, 0, stream>>>(frm, oth, Fm, sp, spec, Vl, U2, U3, br, alp, out);
}
